// QTransformerEncoderLayer_1752346657403
// MI455X (gfx1250) — matrix operand order repaired and checked by reading, not compiled, not run on hardware
//
#include <hip/hip_runtime.h>
#include <hip/hip_bf16.h>

// ---------------------------------------------------------------------------
// QuantumComplex transformer encoder layer, MI455X (gfx1250, wave32, WMMA).
// All GEMM work is done with v_wmma_f32_16x16x32_bf16; f32 only in epilogues.
// ---------------------------------------------------------------------------

typedef __bf16 bf16;
typedef __attribute__((ext_vector_type(16))) __bf16 v16bf;
typedef __attribute__((ext_vector_type(8)))  __bf16 v8bf;
typedef __attribute__((ext_vector_type(8)))  float  v8f;

#define WMMA_BF16(A, Bm, C) \
  __builtin_amdgcn_wmma_f32_16x16x32_bf16(false, (A), false, (Bm), (short)0, (C), false, false)

#define B_  4
#define S_  1024
#define D_  1024
#define H_  16
#define HD_ 64
#define FF_ 4096
#define ND_ ((long)B_ * S_ * D_)   // 4,194,304 elements per [B,S,D] tensor

__device__ __forceinline__ bf16 tobf(float f) {
  union { __hip_bfloat16 h; bf16 b; } u;
  u.h = __float2bfloat16(f);
  return u.b;
}

// A-operand (16xK, 16-bit) fragment: lane<16 holds K {kb..kb+7, kb+16..kb+23},
// lane>=16 holds K {kb+8..kb+15, kb+24..kb+31}. Two 16B chunks per lane.
__device__ __forceinline__ v16bf load_a_frag(const bf16* base, long row, long ldk,
                                             int kb, int hf) {
  const bf16* p = base + row * ldk + kb + hf * 8;
  union { v16bf v; v8bf h[2]; } u;
  u.h[0] = *(const v8bf*)(p);
  u.h[1] = *(const v8bf*)(p + 16);
  return u.v;
}

// B-operand (Kx16, 16-bit) fragment: lane = column; lane<16 holds K {kb..kb+7, kb+16..kb+23},
// lane>=16 holds K {kb+8..kb+15, kb+24..kb+31}. Two 16B chunks.
__device__ __forceinline__ v16bf load_b_frag(const bf16* base, long col, long ldk,
                                             int kb, int hf) {
  const bf16* p = base + col * ldk + kb + hf * 8;
  union { v16bf v; v8bf h[2]; } u;
  u.h[0] = *(const v8bf*)(p);
  u.h[1] = *(const v8bf*)(p + 16);
  return u.v;
}

// ---------------------------------------------------------------------------
__global__ void k_f32_to_bf16(const float* __restrict__ in, bf16* __restrict__ out,
                              long n) {
  long i  = (long)blockIdx.x * blockDim.x + threadIdx.x;
  long st = (long)gridDim.x * blockDim.x;
  for (; i < n; i += st) out[i] = tobf(in[i]);
}

// RMSNorm over last dim (D): nv = rsqrt(mean(xa^2+xb^2)+eps); out bf16.
__global__ void k_rmsnorm(const float* __restrict__ xa, const float* __restrict__ xb,
                          const float* __restrict__ g,
                          bf16* __restrict__ na, bf16* __restrict__ nb) {
  __shared__ float red[256];
  long row = blockIdx.x;
  int  t   = threadIdx.x;
  const float* pa = xa + row * D_;
  const float* pb = xb + row * D_;
  float s = 0.f;
  for (int c = t; c < D_; c += 256) { float a = pa[c], b = pb[c]; s += a * a + b * b; }
  red[t] = s; __syncthreads();
  for (int o = 128; o > 0; o >>= 1) { if (t < o) red[t] += red[t + o]; __syncthreads(); }
  float nv = rsqrtf(red[0] * (1.f / D_) + 1e-6f);
  for (int c = t; c < D_; c += 256) {
    float gg = g[c];
    na[row * D_ + c] = tobf(pa[c] * nv * gg);
    nb[row * D_ + c] = tobf(pb[c] * nv * gg);
  }
}

// qc_dense: Oa = Aa Wa^T + j2 * Ab Wb^T + ba ; Ob = Aa Wb^T + Ab Wa^T + bb.
// Wave computes a 16x32 output tile; 8 WMMAs per K=32 step.
__global__ void k_qc_gemm(const bf16* __restrict__ Aa, const bf16* __restrict__ Ab,
                          const bf16* __restrict__ Wa, const bf16* __restrict__ Wb,
                          const float* __restrict__ ba, const float* __restrict__ bb,
                          const float* __restrict__ theta,
                          float* __restrict__ Oa, float* __restrict__ Ob,
                          int M, int N, int K) {
  int lane  = threadIdx.x & 31;
  int wid   = (int)((blockIdx.x * blockDim.x + threadIdx.x) >> 5);
  int npair = N >> 5;
  int mt = wid / npair, np = wid % npair;
  int hf = lane >> 4, cl = lane & 15;
  long arow = (long)mt * 16 + cl;
  long n0   = (long)np * 32;
  v8f acc[2][4];
#pragma unroll
  for (int t = 0; t < 2; ++t) {
    v8f z = {};
    acc[t][0] = z; acc[t][1] = z; acc[t][2] = z; acc[t][3] = z;
  }
  for (int kb = 0; kb < K; kb += 32) {
    v16bf fa = load_a_frag(Aa, arow, K, kb, hf);
    v16bf fb = load_a_frag(Ab, arow, K, kb, hf);
#pragma unroll
    for (int t = 0; t < 2; ++t) {
      v16bf wfa = load_b_frag(Wa, n0 + t * 16 + cl, K, kb, hf);
      v16bf wfb = load_b_frag(Wb, n0 + t * 16 + cl, K, kb, hf);
      acc[t][0] = WMMA_BF16(fa, wfa, acc[t][0]);   // Aa*Wa
      acc[t][1] = WMMA_BF16(fb, wfb, acc[t][1]);   // Ab*Wb
      acc[t][2] = WMMA_BF16(fa, wfb, acc[t][2]);   // Aa*Wb
      acc[t][3] = WMMA_BF16(fb, wfa, acc[t][3]);   // Ab*Wa
    }
  }
  float j2 = __sinf(2.f * theta[0]) - 1.f;
#pragma unroll
  for (int t = 0; t < 2; ++t) {
    long col = n0 + t * 16 + cl;
    float biasA = ba[col], biasB = bb[col];
#pragma unroll
    for (int r = 0; r < 8; ++r) {
      long row = (long)mt * 16 + r + 8 * hf;      // C-layout: VGPR r <-> M=r+8*half
      Oa[row * N + col] = acc[t][0][r] + j2 * acc[t][1][r] + biasA;
      Ob[row * N + col] = acc[t][2][r] + acc[t][3][r] + biasB;
    }
  }
}

// RoPE (elementwise complex rotation on (a,b)) + pack q,k -> [B,H,S,HD] bf16,
// v -> transposed [B,H,HD,S] bf16 so P*V B-operand loads are contiguous.
__global__ void k_rope_pack(const float* __restrict__ qa, const float* __restrict__ qb,
                            const float* __restrict__ ka, const float* __restrict__ kb,
                            const float* __restrict__ va, const float* __restrict__ vb,
                            const float* __restrict__ rc, const float* __restrict__ rs,
                            bf16* qpa, bf16* qpb, bf16* kpa, bf16* kpb,
                            bf16* vta, bf16* vtb) {
  long i  = (long)blockIdx.x * blockDim.x + threadIdx.x;
  long st = (long)gridDim.x * blockDim.x;
  for (; i < ND_; i += st) {
    int  col = (int)(i % D_);
    long row = i / D_;
    int  s = (int)(row % S_);
    int  b = (int)(row / S_);
    int  h = col / HD_, hd = col % HD_;
    float c = rc[s * HD_ + hd], sn = rs[s * HD_ + hd];
    long bh   = (long)b * H_ + h;
    long pidx = (bh * S_ + s) * HD_ + hd;
    float a0 = qa[i], b0 = qb[i];
    qpa[pidx] = tobf(a0 * c - b0 * sn);
    qpb[pidx] = tobf(a0 * sn + b0 * c);
    float a1 = ka[i], b1 = kb[i];
    kpa[pidx] = tobf(a1 * c - b1 * sn);
    kpb[pidx] = tobf(a1 * sn + b1 * c);
    long vidx = (bh * HD_ + hd) * S_ + s;
    vta[vidx] = tobf(va[i]);
    vtb[vidx] = tobf(vb[i]);
  }
}

// Flash-style attention: one wave per (b,h,16 q-rows); online softmax over
// mag = sqrt(sa^2+sb^2+1e-8)/sqrt(HD) with sa = qa.ka + j2h*qb.kb, sb = qa.kb+qb.ka.
__global__ void k_qc_attention(const bf16* __restrict__ qpa, const bf16* __restrict__ qpb,
                               const bf16* __restrict__ kpa, const bf16* __restrict__ kpb,
                               const bf16* __restrict__ vta, const bf16* __restrict__ vtb,
                               const int* __restrict__ mask,
                               const float* __restrict__ thetas,
                               bf16* __restrict__ outA, bf16* __restrict__ outB) {
  __shared__ __align__(16) bf16 psh[4][16 * 32];   // per-wave P staging (C->A transpose)
  int lane  = threadIdx.x & 31;
  int wslot = threadIdx.x >> 5;
  int wid = blockIdx.x * 4 + wslot;   // over B*H*(S/16) = 4096 waves
  int qt = wid & 63;
  int bh = wid >> 6;
  int h  = bh & (H_ - 1);
  int b  = bh >> 4;
  int hf = lane >> 4, cl = lane & 15;
  const bf16* qA = qpa + (long)bh * S_ * HD_;
  const bf16* qB = qpb + (long)bh * S_ * HD_;
  const bf16* kA = kpa + (long)bh * S_ * HD_;
  const bf16* kB = kpb + (long)bh * S_ * HD_;
  const bf16* vA = vta + (long)bh * HD_ * S_;
  const bf16* vB = vtb + (long)bh * HD_ * S_;
  long qrowA = (long)qt * 16 + cl;
  v16bf Qa[2], Qb[2];
#pragma unroll
  for (int c = 0; c < 2; ++c) {
    Qa[c] = load_a_frag(qA, qrowA, HD_, c * 32, hf);
    Qb[c] = load_a_frag(qB, qrowA, HD_, c * 32, hf);
  }
  float j2h = __sinf(2.f * thetas[h]) - 1.f;
  v8f zero8 = {};
  v8f Oacc[4] = {zero8, zero8, zero8, zero8};
  v8f Obcc[4] = {zero8, zero8, zero8, zero8};
  float mrow[8], lrow[8];
#pragma unroll
  for (int r = 0; r < 8; ++r) { mrow[r] = -1e30f; lrow[r] = 0.f; }

  for (int k0 = 0; k0 < S_; k0 += 32) {
    float mg[2][8];
#pragma unroll
    for (int t = 0; t < 2; ++t) {
      v8f sAA = {}, sBB = {}, sAB = {}, sBA = {};
      long kcolA = (long)k0 + t * 16 + cl;
#pragma unroll
      for (int c = 0; c < 2; ++c) {
        v16bf Kfa = load_b_frag(kA, kcolA, HD_, c * 32, hf);
        v16bf Kfb = load_b_frag(kB, kcolA, HD_, c * 32, hf);
        sAA = WMMA_BF16(Qa[c], Kfa, sAA);
        sBB = WMMA_BF16(Qb[c], Kfb, sBB);
        sAB = WMMA_BF16(Qa[c], Kfb, sAB);
        sBA = WMMA_BF16(Qb[c], Kfa, sBA);
      }
#pragma unroll
      for (int r = 0; r < 8; ++r) {
        float sa  = sAA[r] + j2h * sBB[r];
        float sb  = sAB[r] + sBA[r];
        float mag = sqrtf(sa * sa + sb * sb + 1e-8f) * 0.125f;   // / sqrt(HD)
        int qrow = qt * 16 + r + 8 * hf;
        int kcol = k0 + t * 16 + cl;
        mg[t][r] = (mask[(long)qrow * S_ + kcol] == 0) ? -1e9f : mag;
      }
    }
    // online softmax row stats: rows live across 16 lanes within a half-wave,
    // so xor-shuffles with masks 1/2/4/8 reduce exactly one row.
#pragma unroll
    for (int r = 0; r < 8; ++r) {
      float v = fmaxf(mg[0][r], mg[1][r]);
      v = fmaxf(v, __shfl_xor(v, 1, 32));
      v = fmaxf(v, __shfl_xor(v, 2, 32));
      v = fmaxf(v, __shfl_xor(v, 4, 32));
      v = fmaxf(v, __shfl_xor(v, 8, 32));
      float nm = fmaxf(mrow[r], v);
      float p0 = __expf(mg[0][r] - nm);
      float p1 = __expf(mg[1][r] - nm);
      mg[0][r] = p0; mg[1][r] = p1;
      float ps = p0 + p1;
      ps += __shfl_xor(ps, 1, 32);
      ps += __shfl_xor(ps, 2, 32);
      ps += __shfl_xor(ps, 4, 32);
      ps += __shfl_xor(ps, 8, 32);
      float sc = __expf(mrow[r] - nm);
      lrow[r] = lrow[r] * sc + ps;
      mrow[r] = nm;
#pragma unroll
      for (int nt = 0; nt < 4; ++nt) { Oacc[nt][r] *= sc; Obcc[nt][r] *= sc; }
    }
    // P: C-layout -> row-major LDS tile -> A-layout fragment
#pragma unroll
    for (int t = 0; t < 2; ++t)
#pragma unroll
      for (int r = 0; r < 8; ++r)
        psh[wslot][(r + 8 * hf) * 32 + t * 16 + cl] = tobf(mg[t][r]);
    __builtin_amdgcn_wave_barrier();
    asm volatile("s_wait_dscnt 0x0" ::: "memory");
    v16bf Pf = load_a_frag(&psh[wslot][0], cl, 32, 0, hf);
#pragma unroll
    for (int nt = 0; nt < 4; ++nt) {
      v16bf Vfa = load_b_frag(vA, nt * 16 + cl, S_, k0, hf);
      v16bf Vfb = load_b_frag(vB, nt * 16 + cl, S_, k0, hf);
      Oacc[nt] = WMMA_BF16(Pf, Vfa, Oacc[nt]);
      Obcc[nt] = WMMA_BF16(Pf, Vfb, Obcc[nt]);
    }
  }
  // normalize and write un-shaped [B*S, D] bf16 (ready for the 'o' projection)
#pragma unroll
  for (int r = 0; r < 8; ++r) {
    float rl = 1.f / lrow[r];
    long orow = (long)b * S_ + qt * 16 + r + 8 * hf;
#pragma unroll
    for (int nt = 0; nt < 4; ++nt) {
      long ocol = (long)h * HD_ + nt * 16 + cl;
      outA[orow * D_ + ocol] = tobf(Oacc[nt][r] * rl);
      outB[orow * D_ + ocol] = tobf(Obcc[nt][r] * rl);
    }
  }
}

// Fused gate+up qc_dense + QC-SwiGLU: Ha = silu(|g|)*ua, Hb = silu(|g|)*ub (bf16).
__global__ void k_qc_ffn(const bf16* __restrict__ Aa, const bf16* __restrict__ Ab,
                         const bf16* __restrict__ GWa, const bf16* __restrict__ GWb,
                         const bf16* __restrict__ UWa, const bf16* __restrict__ UWb,
                         const float* __restrict__ gba, const float* __restrict__ gbb,
                         const float* __restrict__ uba, const float* __restrict__ ubb,
                         const float* __restrict__ theta,
                         bf16* __restrict__ Ha, bf16* __restrict__ Hb,
                         int M, int N, int K) {
  int lane   = threadIdx.x & 31;
  int wid    = (int)((blockIdx.x * blockDim.x + threadIdx.x) >> 5);
  int ntiles = N >> 4;
  int mt = wid / ntiles, nt = wid % ntiles;
  int hf = lane >> 4, cl = lane & 15;
  long arow = (long)mt * 16 + cl;
  long n0   = (long)nt * 16;
  v8f gAA = {}, gBB = {}, gAB = {}, gBA = {};
  v8f uAA = {}, uBB = {}, uAB = {}, uBA = {};
  for (int kb = 0; kb < K; kb += 32) {
    v16bf fa  = load_a_frag(Aa, arow, K, kb, hf);
    v16bf fb  = load_a_frag(Ab, arow, K, kb, hf);
    v16bf gwa = load_b_frag(GWa, n0 + cl, K, kb, hf);
    v16bf gwb = load_b_frag(GWb, n0 + cl, K, kb, hf);
    v16bf uwa = load_b_frag(UWa, n0 + cl, K, kb, hf);
    v16bf uwb = load_b_frag(UWb, n0 + cl, K, kb, hf);
    gAA = WMMA_BF16(fa, gwa, gAA); gBB = WMMA_BF16(fb, gwb, gBB);
    gAB = WMMA_BF16(fa, gwb, gAB); gBA = WMMA_BF16(fb, gwa, gBA);
    uAA = WMMA_BF16(fa, uwa, uAA); uBB = WMMA_BF16(fb, uwb, uBB);
    uAB = WMMA_BF16(fa, uwb, uAB); uBA = WMMA_BF16(fb, uwa, uBA);
  }
  float j2  = __sinf(2.f * theta[0]) - 1.f;
  long col  = n0 + cl;
  float gbA = gba[col], gbB = gbb[col], ubA = uba[col], ubB = ubb[col];
#pragma unroll
  for (int r = 0; r < 8; ++r) {
    long row = (long)mt * 16 + r + 8 * hf;
    float ga = gAA[r] + j2 * gBB[r] + gbA;
    float gb = gAB[r] + gBA[r] + gbB;
    float ua = uAA[r] + j2 * uBB[r] + ubA;
    float ub = uAB[r] + uBA[r] + ubB;
    float x  = sqrtf(ga * ga + gb * gb);
    float gm = x / (1.f + __expf(-x));     // silu
    Ha[row * N + col] = tobf(gm * ua);
    Hb[row * N + col] = tobf(gm * ub);
  }
}

__global__ void k_add(const float* __restrict__ a, const float* __restrict__ b,
                      float* __restrict__ c, long n) {
  long i  = (long)blockIdx.x * blockDim.x + threadIdx.x;
  long st = (long)gridDim.x * blockDim.x;
  for (; i < n; i += st) c[i] = a[i] + b[i];
}

// ---------------------------------------------------------------------------
extern "C" void kernel_launch(void* const* d_in, const int* in_sizes, int n_in,
                              void* d_out, int out_size, void* d_ws, size_t ws_size,
                              hipStream_t stream) {
  (void)in_sizes; (void)n_in; (void)out_size; (void)ws_size;
  // Flat input order = JAX tree_leaves of setup_inputs() (dict keys sorted).
  enum {
    IN_MASK = 0, IN_G1, IN_G2,
    IN_GATE_BA, IN_GATE_BB, IN_GATE_WA, IN_GATE_WB,
    IN_K_BA, IN_K_BB, IN_K_WA, IN_K_WB,
    IN_LT,
    IN_O_BA, IN_O_BB, IN_O_WA, IN_O_WB,
    IN_OUT_BA, IN_OUT_BB, IN_OUT_WA, IN_OUT_WB,
    IN_Q_BA, IN_Q_BB, IN_Q_WA, IN_Q_WB,
    IN_TH,
    IN_UP_BA, IN_UP_BB, IN_UP_WA, IN_UP_WB,
    IN_V_BA, IN_V_BB, IN_V_WA, IN_V_WB,
    IN_RC, IN_RS, IN_XA, IN_XB
  };
  const int*   mask = (const int*)d_in[IN_MASK];
  const float* g1   = (const float*)d_in[IN_G1];
  const float* g2   = (const float*)d_in[IN_G2];
  const float* lt   = (const float*)d_in[IN_LT];
  const float* th   = (const float*)d_in[IN_TH];
  const float* rc   = (const float*)d_in[IN_RC];
  const float* rs   = (const float*)d_in[IN_RS];
  const float* xa   = (const float*)d_in[IN_XA];
  const float* xb   = (const float*)d_in[IN_XB];

  char* ws = (char*)d_ws;
  const size_t MB = 1024ull * 1024ull;
  // bf16 weights (64 MB)
  bf16* qwa = (bf16*)(ws + 0  * MB); bf16* qwb = (bf16*)(ws + 2  * MB);
  bf16* kwa = (bf16*)(ws + 4  * MB); bf16* kwb = (bf16*)(ws + 6  * MB);
  bf16* vwa = (bf16*)(ws + 8  * MB); bf16* vwb = (bf16*)(ws + 10 * MB);
  bf16* owa = (bf16*)(ws + 12 * MB); bf16* owb = (bf16*)(ws + 14 * MB);
  bf16* gwa = (bf16*)(ws + 16 * MB); bf16* gwb = (bf16*)(ws + 24 * MB);
  bf16* uwa = (bf16*)(ws + 32 * MB); bf16* uwb = (bf16*)(ws + 40 * MB);
  bf16* fwa = (bf16*)(ws + 48 * MB); bf16* fwb = (bf16*)(ws + 56 * MB);
  // normalized activations bf16 (reused for both norms)
  bf16* na = (bf16*)(ws + 64 * MB);  bf16* nb = (bf16*)(ws + 72 * MB);
  // phase A: QKV f32 outputs (80..176 MB)
  float* qaF = (float*)(ws + 80  * MB); float* qbF = (float*)(ws + 96  * MB);
  float* kaF = (float*)(ws + 112 * MB); float* kbF = (float*)(ws + 128 * MB);
  float* vaF = (float*)(ws + 144 * MB); float* vbF = (float*)(ws + 160 * MB);
  // phase B+ (overlaps phase A after it is dead)
  bf16*  attA = (bf16*)(ws + 80  * MB); bf16*  attB = (bf16*)(ws + 88  * MB);
  float* opA  = (float*)(ws + 96  * MB); float* opB  = (float*)(ws + 112 * MB);
  float* xa2  = (float*)(ws + 128 * MB); float* xb2  = (float*)(ws + 144 * MB);
  bf16*  ha   = (bf16*)(ws + 160 * MB); bf16*  hb   = (bf16*)(ws + 168 * MB);
  float* faF  = (float*)(ws + 96  * MB); float* fbF  = (float*)(ws + 112 * MB);
  // packed q/k [B,H,S,HD] and vT [B,H,HD,S] bf16 (176..224 MB)
  bf16* qpa = (bf16*)(ws + 176 * MB); bf16* qpb = (bf16*)(ws + 184 * MB);
  bf16* kpa = (bf16*)(ws + 192 * MB); bf16* kpb = (bf16*)(ws + 200 * MB);
  bf16* vta = (bf16*)(ws + 208 * MB); bf16* vtb = (bf16*)(ws + 216 * MB);

  // 1. weights f32 -> bf16
  struct CV { const void* src; bf16* dst; long n; };
  const long DD = (long)D_ * D_, DF = (long)D_ * FF_;
  CV cv[16] = {
    {d_in[IN_Q_WA],   qwa, DD}, {d_in[IN_Q_WB],   qwb, DD},
    {d_in[IN_K_WA],   kwa, DD}, {d_in[IN_K_WB],   kwb, DD},
    {d_in[IN_V_WA],   vwa, DD}, {d_in[IN_V_WB],   vwb, DD},
    {d_in[IN_O_WA],   owa, DD}, {d_in[IN_O_WB],   owb, DD},
    {d_in[IN_GATE_WA], gwa, DF}, {d_in[IN_GATE_WB], gwb, DF},
    {d_in[IN_UP_WA],   uwa, DF}, {d_in[IN_UP_WB],   uwb, DF},
    {d_in[IN_OUT_WA],  fwa, DF}, {d_in[IN_OUT_WB],  fwb, DF},
    {d_in[IN_XA], nullptr, 0}, {d_in[IN_XB], nullptr, 0}
  };
  for (int i = 0; i < 14; ++i)
    k_f32_to_bf16<<<4096, 256, 0, stream>>>((const float*)cv[i].src, cv[i].dst, cv[i].n);

  // 2. pre-norm 1
  k_rmsnorm<<<B_ * S_, 256, 0, stream>>>(xa, xb, g1, na, nb);

  // 3. Q,K,V projections (waves = (4096/16)*(1024/32) = 8192 -> 1024 blocks)
  k_qc_gemm<<<1024, 256, 0, stream>>>(na, nb, qwa, qwb,
      (const float*)d_in[IN_Q_BA], (const float*)d_in[IN_Q_BB], lt,
      qaF, qbF, B_ * S_, D_, D_);
  k_qc_gemm<<<1024, 256, 0, stream>>>(na, nb, kwa, kwb,
      (const float*)d_in[IN_K_BA], (const float*)d_in[IN_K_BB], lt,
      kaF, kbF, B_ * S_, D_, D_);
  k_qc_gemm<<<1024, 256, 0, stream>>>(na, nb, vwa, vwb,
      (const float*)d_in[IN_V_BA], (const float*)d_in[IN_V_BB], lt,
      vaF, vbF, B_ * S_, D_, D_);

  // 4. RoPE + head pack (q,k) and v transpose
  k_rope_pack<<<4096, 256, 0, stream>>>(qaF, qbF, kaF, kbF, vaF, vbF, rc, rs,
                                        qpa, qpb, kpa, kpb, vta, vtb);

  // 5. attention (4096 waves, 4 per block)
  k_qc_attention<<<1024, 128, 0, stream>>>(qpa, qpb, kpa, kpb, vta, vtb,
                                           mask, th, attA, attB);

  // 6. output projection
  k_qc_gemm<<<1024, 256, 0, stream>>>(attA, attB, owa, owb,
      (const float*)d_in[IN_O_BA], (const float*)d_in[IN_O_BB], lt,
      opA, opB, B_ * S_, D_, D_);

  // 7. residual 1
  k_add<<<16384, 256, 0, stream>>>(xa, opA, xa2, ND_);
  k_add<<<16384, 256, 0, stream>>>(xb, opB, xb2, ND_);

  // 8. pre-norm 2
  k_rmsnorm<<<B_ * S_, 256, 0, stream>>>(xa2, xb2, g2, na, nb);

  // 9. fused gate+up + QC-SwiGLU (waves = 256*256 = 65536 -> 8192 blocks)
  k_qc_ffn<<<8192, 256, 0, stream>>>(na, nb, gwa, gwb, uwa, uwb,
      (const float*)d_in[IN_GATE_BA], (const float*)d_in[IN_GATE_BB],
      (const float*)d_in[IN_UP_BA],   (const float*)d_in[IN_UP_BB], lt,
      ha, hb, B_ * S_, FF_, D_);

  // 10. FFN output projection (K = FF)
  k_qc_gemm<<<1024, 256, 0, stream>>>(ha, hb, fwa, fwb,
      (const float*)d_in[IN_OUT_BA], (const float*)d_in[IN_OUT_BB], lt,
      faF, fbF, B_ * S_, D_, FF_);

  // 11. residual 2 -> d_out [2,B,S,D] f32
  float* out = (float*)d_out;
  k_add<<<16384, 256, 0, stream>>>(xa2, faF, out, ND_);
  k_add<<<16384, 256, 0, stream>>>(xb2, fbF, out + ND_, ND_);
}
